// DeformTransWorldFeat_aio_68393059221979
// MI455X (gfx1250) — compile-verified, matrix operand order checked by reading, not run on hardware
//
#include <hip/hip_runtime.h>
#include <math.h>

// ---------------------------------------------------------------------------
// MI455X (gfx1250, wave32) deformable-transformer encoder.
// GEMMs: v_wmma_f32_16x16x32_bf16, block tile 64x128, wave tile 16x64
// (4 accumulators reuse one A fragment -> 2.5 b128 loads per WMMA).
// ---------------------------------------------------------------------------

typedef __bf16 bf16_t;
typedef __attribute__((ext_vector_type(16))) __bf16 v16bf;
typedef __attribute__((ext_vector_type(8)))  __bf16 v8bf;
typedef __attribute__((ext_vector_type(8)))  float  v8f;

#define D_MODEL 128
#define NPTS_C  4

enum { EPI_NONE = 0, EPI_RELU = 1, EPI_LN = 2, EPI_RELU_T = 3 };

// Load one 16x32 WMMA operand fragment (A row-major / B^T row-major):
// lane holds two contiguous 8 x bf16 (16B) runs selected by lane>>4.
__device__ __forceinline__ v16bf ldfrag(const bf16_t* p)
{
  v8bf lo = *(const v8bf*)p;
  v8bf hi = *(const v8bf*)(p + 16);
  return __builtin_shufflevector(lo, hi, 0,1,2,3,4,5,6,7,8,9,10,11,12,13,14,15);
}

// ---------------------------------------------------------------------------
// C[M,N] = A[M,K](bf16) * Bt[Npad,K](bf16)^T + bias, with epilogue.
// Bt is zero-padded to Npad = ceil64(N) rows so the 4-tile inner loop needs
// no load guards. Block = 256 threads = 8 waves = 4 row-groups x 2 col-groups.
// ---------------------------------------------------------------------------
__global__ __launch_bounds__(256) void gemm_bf16_wmma(
    const bf16_t* __restrict__ A,    // [M,K] bf16 row-major
    const bf16_t* __restrict__ Bt,   // [Npad,K] bf16 row-major (weights^T)
    const float*  __restrict__ bias, // [N] (may be null)
    const float*  __restrict__ res,  // [M,N] fp32 residual (LN only)
    const float*  __restrict__ ln_s, // [N] LN scale (LN only)
    const float*  __restrict__ ln_b, // [N] LN bias  (LN only)
    float*        __restrict__ Cf,   // fp32 out (may be null)
    bf16_t*       __restrict__ Cb,   // bf16 out (may be null)
    int M, int N, int K, int epi)
{
  __shared__ float s_tile[64][132];   // LN staging (33 KB of 320 KB WGP LDS)
  __shared__ float s_mu[64];
  __shared__ float s_rs[64];

  const int wave = threadIdx.x >> 5;
  const int lane = threadIdx.x & 31;
  const int lcol = lane & 15;         // A-frag row / B-frag col / C col
  const int lhi  = lane >> 4;         // K half-select
  const int wrow = wave >> 1;         // 0..3 : 16-row group
  const int wcol = (wave & 1) << 6;   // 0/64 : 64-col group
  const int m0    = (blockIdx.x << 6) + (wrow << 4);
  const int nbase = (blockIdx.y << 7) + wcol;

  v8f acc[4] = {};
  const bool active = (nbase < N);    // wave-uniform
  if (active) {
    const bf16_t* aptr = A  + (size_t)(m0 + lcol) * K + (lhi << 3);
    const bf16_t* bptr = Bt + (size_t)(nbase + lcol) * K + (lhi << 3);
    const size_t bs = (size_t)16 * K;  // stride between 16-col B tiles
    for (int k0 = 0; k0 < K; k0 += 32) {
      __builtin_prefetch(aptr + k0 + 256, 0, 1);   // global_prefetch_b8
      v16bf av  = ldfrag(aptr + k0);
      v16bf bv0 = ldfrag(bptr + k0);
      v16bf bv1 = ldfrag(bptr + bs + k0);
      v16bf bv2 = ldfrag(bptr + 2 * bs + k0);
      v16bf bv3 = ldfrag(bptr + 3 * bs + k0);
      acc[0] = __builtin_amdgcn_wmma_f32_16x16x32_bf16(false, av, false, bv0,
                                                       (short)0, acc[0], false, false);
      acc[1] = __builtin_amdgcn_wmma_f32_16x16x32_bf16(false, av, false, bv1,
                                                       (short)0, acc[1], false, false);
      acc[2] = __builtin_amdgcn_wmma_f32_16x16x32_bf16(false, av, false, bv2,
                                                       (short)0, acc[2], false, false);
      acc[3] = __builtin_amdgcn_wmma_f32_16x16x32_bf16(false, av, false, bv3,
                                                       (short)0, acc[3], false, false);
    }
  }

  if (epi == EPI_LN) {
    // N == 128 here: block owns 64 complete rows -> row-wise LayerNorm.
#pragma unroll
    for (int t = 0; t < 4; ++t) {
      const int n = nbase + (t << 4) + lcol;
      const float bcol = bias[n];
#pragma unroll
      for (int v = 0; v < 8; ++v) {
        const int lr = (wrow << 4) + v + (lhi << 3);
        s_tile[lr][wcol + (t << 4) + lcol] =
            acc[t][v] + bcol + res[(size_t)((blockIdx.x << 6) + lr) * N + n];
      }
    }
    __syncthreads();
    if (threadIdx.x < 64) {
      float s = 0.0f, s2 = 0.0f;
      for (int c = 0; c < 128; ++c) {
        const float t = s_tile[threadIdx.x][c];
        s += t; s2 += t * t;
      }
      const float mu  = s * (1.0f / 128.0f);
      const float var = s2 * (1.0f / 128.0f) - mu * mu;
      s_mu[threadIdx.x] = mu;
      s_rs[threadIdx.x] = rsqrtf(var + 1e-5f);
    }
    __syncthreads();
#pragma unroll
    for (int j = 0; j < 32; ++j) {    // 64*128 / 256 threads
      const int e = threadIdx.x * 32 + j;
      const int r = e >> 7, c = e & 127;
      const float o = (s_tile[r][c] - s_mu[r]) * s_rs[r] * ln_s[c] + ln_b[c];
      const size_t idx = (size_t)((blockIdx.x << 6) + r) * N + c;
      Cf[idx] = o;
      Cb[idx] = (bf16_t)o;
    }
  } else if (active) {
#pragma unroll
    for (int t = 0; t < 4; ++t) {
      const int n = nbase + (t << 4) + lcol;
      const int n0t = nbase + (t << 4);
      if (n0t < N) {                  // store guard (logical N)
        const float bcol = bias ? bias[n] : 0.0f;
#pragma unroll
        for (int v = 0; v < 8; ++v) {
          const int m = m0 + v + (lhi << 3);
          float o = acc[t][v] + bcol;
          if (epi != EPI_NONE) o = fmaxf(o, 0.0f);
          if (epi == EPI_RELU_T) {
            Cf[(size_t)n * M + m] = o;       // [N, M] channel-first output
          } else {
            const size_t idx = (size_t)m * N + n;
            if (Cf) Cf[idx] = o;
            if (Cb) Cb[idx] = (bf16_t)o;
          }
        }
      }
    }
  }
}

// ---------------------------------------------------------------------------
// Pack weights to bf16 transposed layout Wt[Npad][K], zero-padding rows >= N.
// trans=1: input W is [K,N] row-major; trans=0: input already [N,K].
// ---------------------------------------------------------------------------
__global__ void pack_wt_kernel(const float* __restrict__ W,
                               bf16_t* __restrict__ Wt,
                               int K, int N, int Npad, int trans)
{
  const int i = blockIdx.x * blockDim.x + threadIdx.x;
  if (i >= Npad * K) return;
  const int n = i / K, k = i - n * K;
  float v = 0.0f;
  if (n < N) v = trans ? W[(size_t)k * N + n] : W[(size_t)n * K + k];
  Wt[i] = (bf16_t)v;
}

// x [512, Q] fp32 (channel-major) -> A [Q, 512] bf16 (row-major GEMM operand)
__global__ void xpose_kernel(const float* __restrict__ x,
                             bf16_t* __restrict__ Abf, int Q)
{
  const int q = blockIdx.x * blockDim.x + threadIdx.x;
  if (q >= Q) return;
  for (int c = 0; c < 512; c += 8) {
    v8bf v;
#pragma unroll
    for (int j = 0; j < 8; ++j) v[j] = (bf16_t)x[(size_t)(c + j) * Q + q];
    *(v8bf*)(Abf + (size_t)q * 512 + c) = v;
  }
}

// Sine/cosine positional embedding (normalize=True, temp=10000, scale=2*pi).
__global__ void pos_embed_kernel(float* __restrict__ pos, int H, int W)
{
  const int q = blockIdx.x;
  const int d = threadIdx.x;           // blockDim = 128
  const int h = q / W, w = q - h * W;
  const float scale = 6.283185307179586f;
  float e; int i;
  if (d < 64) { e = (float)(h + 1) / ((float)H + 1e-6f) * scale; i = d; }
  else        { e = (float)(w + 1) / ((float)W + 1e-6f) * scale; i = d - 64; }
  const float dim_t = powf(10000.0f, (float)(2 * (i >> 1)) * (1.0f / 64.0f));
  const float p = e / dim_t;
  pos[(size_t)q * D_MODEL + d] = (i & 1) ? cosf(p) : sinf(p);
}

// q = bf16(src + pos)
__global__ void qprep_kernel(const float* __restrict__ src,
                             const float* __restrict__ pos,
                             bf16_t* __restrict__ qb, int n)
{
  const int i = blockIdx.x * blockDim.x + threadIdx.x;
  if (i < n) qb[i] = (bf16_t)(src[i] + pos[i]);
}

// ---------------------------------------------------------------------------
// Deformable bilinear sampling + softmax attention weighting.
// 2 queries per 256-thread block; thread = (head, channel) of one query.
// ---------------------------------------------------------------------------
__global__ __launch_bounds__(256) void deform_sample_kernel(
    const float* __restrict__ off_raw, const float* __restrict__ off_b,
    const float* __restrict__ aw_raw,  const float* __restrict__ aw_b,
    const float* __restrict__ val,     bf16_t* __restrict__ attn_bf,
    int H, int W)
{
  const int Q = H * W;
  const int q = blockIdx.x * 2 + (threadIdx.x >> 7);
  if (q >= Q) return;
  const int r = threadIdx.x & 127;
  const int h = r >> 4;
  const int c = r & 15;
  const int qy = q / W, qx = q - qy * W;

  // softmax over the 4 points of this head
  float a[NPTS_C];
  float amax = -1e30f;
#pragma unroll
  for (int p = 0; p < NPTS_C; ++p) {
    a[p] = aw_raw[(size_t)q * 32 + h * 4 + p] + aw_b[h * 4 + p];
    amax = fmaxf(amax, a[p]);
  }
  float asum = 0.0f;
#pragma unroll
  for (int p = 0; p < NPTS_C; ++p) { a[p] = expf(a[p] - amax); asum += a[p]; }
  const float inv = 1.0f / asum;

  float accv = 0.0f;
#pragma unroll
  for (int p = 0; p < NPTS_C; ++p) {
    const float ox = off_raw[(size_t)q * 64 + h * 8 + p * 2 + 0] + off_b[h * 8 + p * 2 + 0];
    const float oy = off_raw[(size_t)q * 64 + h * 8 + p * 2 + 1] + off_b[h * 8 + p * 2 + 1];
    // loc = ref + off/(W,H); x = loc.x*W - 0.5 = qx + ox (refs at cell centers)
    const float x = (float)qx + ox;
    const float y = (float)qy + oy;
    const float x0f = floorf(x), y0f = floorf(y);
    const float wx1 = x - x0f,  wy1 = y - y0f;
    const int x0 = (int)x0f, y0 = (int)y0f;
    float g = 0.0f;
#pragma unroll
    for (int dy = 0; dy < 2; ++dy) {
#pragma unroll
      for (int dx = 0; dx < 2; ++dx) {
        const int xi = x0 + dx, yi = y0 + dy;
        if (xi >= 0 && xi < W && yi >= 0 && yi < H) {   // zero padding
          const float wgt = (dx ? wx1 : 1.0f - wx1) * (dy ? wy1 : 1.0f - wy1);
          g += wgt * val[((size_t)yi * W + xi) * D_MODEL + h * 16 + c];
        }
      }
    }
    accv += a[p] * inv * g;
  }
  attn_bf[(size_t)q * D_MODEL + r] = (bf16_t)accv;
}

// ---------------------------------------------------------------------------
extern "C" void kernel_launch(void* const* d_in, const int* in_sizes, int n_in,
                              void* d_out, int out_size, void* d_ws, size_t ws_size,
                              hipStream_t stream)
{
  (void)in_sizes; (void)n_in; (void)out_size; (void)ws_size;
  const int H = 120, W = 360, Q = H * W;
  const int D = 128, Dff = 512, L = 3, NC = 512;

  const float* x       = (const float*)d_in[0];
  const float* merge_w = (const float*)d_in[1];
  const float* merge_b = (const float*)d_in[2];
  const float* off_w   = (const float*)d_in[3];
  const float* off_b   = (const float*)d_in[4];
  const float* aw_w    = (const float*)d_in[5];
  const float* aw_b    = (const float*)d_in[6];
  const float* val_w   = (const float*)d_in[7];
  const float* val_b   = (const float*)d_in[8];
  const float* out_w   = (const float*)d_in[9];
  const float* out_b   = (const float*)d_in[10];
  const float* n1_s    = (const float*)d_in[11];
  const float* n1_b    = (const float*)d_in[12];
  const float* l1_w    = (const float*)d_in[13];
  const float* l1_b    = (const float*)d_in[14];
  const float* l2_w    = (const float*)d_in[15];
  const float* l2_b    = (const float*)d_in[16];
  const float* n2_s    = (const float*)d_in[17];
  const float* n2_b    = (const float*)d_in[18];
  const float* outc_w  = (const float*)d_in[19];
  const float* outc_b  = (const float*)d_in[20];
  float* out = (float*)d_out;

  // --- workspace carve-out (256B aligned) ---
  char* base = (char*)d_ws;
  size_t cur = 0;
  auto carve = [&](size_t bytes) -> char* {
    char* p = base + cur;
    cur += (bytes + 255) & ~(size_t)255;
    return p;
  };
  bf16_t* Abf     = (bf16_t*)carve((size_t)Q * NC * 2);
  float*  pos     = (float* )carve((size_t)Q * D * 4);
  float*  srcA    = (float* )carve((size_t)Q * D * 4);
  float*  srcB    = (float* )carve((size_t)Q * D * 4);
  bf16_t* src_bf  = (bf16_t*)carve((size_t)Q * D * 2);
  bf16_t* q_bf    = (bf16_t*)carve((size_t)Q * D * 2);
  float*  off_raw = (float* )carve((size_t)Q * 64 * 4);
  float*  aw_raw  = (float* )carve((size_t)Q * 32 * 4);
  float*  valbuf  = (float* )carve((size_t)Q * D * 4);
  bf16_t* attn_bf = (bf16_t*)carve((size_t)Q * D * 2);
  bf16_t* ff_bf   = (bf16_t*)carve((size_t)Q * Dff * 2);
  bf16_t* wt_merge = (bf16_t*)carve((size_t)D * NC * 2);
  bf16_t* wt_off   = (bf16_t*)carve((size_t)L * 64 * D * 2);
  bf16_t* wt_aw    = (bf16_t*)carve((size_t)L * 64 * D * 2);  // padded 32->64
  bf16_t* wt_val   = (bf16_t*)carve((size_t)L * D * D * 2);
  bf16_t* wt_out   = (bf16_t*)carve((size_t)L * D * D * 2);
  bf16_t* wt_l1    = (bf16_t*)carve((size_t)L * Dff * D * 2);
  bf16_t* wt_l2    = (bf16_t*)carve((size_t)L * D * Dff * 2);
  bf16_t* wt_outc  = (bf16_t*)carve((size_t)D * D * 2);

  auto pack = [&](const float* Wsrc, bf16_t* Wt, int K, int N, int trans) {
    const int Npad = (N + 63) & ~63;
    const int n = K * Npad;
    pack_wt_kernel<<<(n + 255) / 256, 256, 0, stream>>>(Wsrc, Wt, K, N, Npad, trans);
  };
  auto gemm = [&](const bf16_t* A, const bf16_t* Bt, const float* bias,
                  const float* res, const float* lns, const float* lnb,
                  float* Cf, bf16_t* Cb, int M, int N, int K, int epi) {
    dim3 grid(M / 64, (N + 127) / 128);
    gemm_bf16_wmma<<<grid, 256, 0, stream>>>(A, Bt, bias, res, lns, lnb,
                                             Cf, Cb, M, N, K, epi);
  };

  // --- weight packing (bf16, transposed to [Npad,K]) ---
  pack(merge_w, wt_merge, NC, D, 0);              // merge_w is [D, NC] = [N,K]
  pack(outc_w,  wt_outc,  D,  D, 0);              // outc_w  is [D, D]  = [N,K]
  for (int l = 0; l < L; ++l) {
    pack(off_w + (size_t)l * D * 64,   wt_off + (size_t)l * 64 * D,  D,   64, 1);
    pack(aw_w  + (size_t)l * D * 32,   wt_aw  + (size_t)l * 64 * D,  D,   32, 1);
    pack(val_w + (size_t)l * D * D,    wt_val + (size_t)l * D * D,   D,   D,  1);
    pack(out_w + (size_t)l * D * D,    wt_out + (size_t)l * D * D,   D,   D,  1);
    pack(l1_w  + (size_t)l * D * Dff,  wt_l1  + (size_t)l * Dff * D, D,   Dff, 1);
    pack(l2_w  + (size_t)l * Dff * D,  wt_l2  + (size_t)l * D * Dff, Dff, D,  1);
  }

  // --- input transpose + positional embedding ---
  xpose_kernel<<<(Q + 255) / 256, 256, 0, stream>>>(x, Abf, Q);
  pos_embed_kernel<<<Q, 128, 0, stream>>>(pos, H, W);

  // --- merge: src = relu(x^T @ merge_w^T + b) ---
  gemm(Abf, wt_merge, merge_b, nullptr, nullptr, nullptr,
       srcA, src_bf, Q, D, NC, EPI_RELU);

  // --- encoder layers ---
  for (int l = 0; l < L; ++l) {
    qprep_kernel<<<(Q * D + 255) / 256, 256, 0, stream>>>(srcA, pos, q_bf, Q * D);

    // biases for off/aw are applied in the sampling kernel (not here)
    gemm(q_bf,   wt_off + (size_t)l * 64 * D, nullptr,
         nullptr, nullptr, nullptr, off_raw, nullptr, Q, 64, D, EPI_NONE);
    gemm(q_bf,   wt_aw  + (size_t)l * 64 * D, nullptr,
         nullptr, nullptr, nullptr, aw_raw, nullptr, Q, 32, D, EPI_NONE);
    gemm(src_bf, wt_val + (size_t)l * D * D,  val_b + (size_t)l * D,
         nullptr, nullptr, nullptr, valbuf, nullptr, Q, D, D, EPI_NONE);

    deform_sample_kernel<<<(Q + 1) / 2, 256, 0, stream>>>(
        off_raw, off_b + (size_t)l * 64, aw_raw, aw_b + (size_t)l * 32,
        valbuf, attn_bf, H, W);

    // out projection + residual + LN1  -> srcB (+ src_bf)
    gemm(attn_bf, wt_out + (size_t)l * D * D, out_b + (size_t)l * D,
         srcA, n1_s + (size_t)l * D, n1_b + (size_t)l * D,
         srcB, src_bf, Q, D, D, EPI_LN);

    // FFN1: relu(srcB @ l1 + b)  (bf16 only, feeds FFN2)
    gemm(src_bf, wt_l1 + (size_t)l * Dff * D, l1_b + (size_t)l * Dff,
         nullptr, nullptr, nullptr, nullptr, ff_bf, Q, Dff, D, EPI_RELU);

    // FFN2 + residual + LN2 -> srcA (+ src_bf)
    gemm(ff_bf, wt_l2 + (size_t)l * D * Dff, l2_b + (size_t)l * D,
         srcB, n2_s + (size_t)l * D, n2_b + (size_t)l * D,
         srcA, src_bf, Q, D, Dff, EPI_LN);
  }

  // --- output conv: relu(src @ outc^T + b), stored channel-first [D, H*W] ---
  gemm(src_bf, wt_outc, outc_b, nullptr, nullptr, nullptr,
       out, nullptr, Q, D, D, EPI_RELU_T);
}
